// SentimentClassifier_50044958933549
// MI455X (gfx1250) — matrix operand order repaired and checked by reading, not compiled, not run on hardware
//
#include <hip/hip_runtime.h>
#include <hip/hip_bf16.h>
#include <cstdint>

// ---------------------------------------------------------------------------
// Transformer sentiment classifier forward, CDNA5 (gfx1250, wave32, WMMA).
// ~1.4 TFLOP/call, weights ~190MB (L2-resident) -> compute-bound: everything
// runs on v_wmma_f32_16x16x32_bf16 (bf16 in, fp32 accumulate).  All GEMM
// operands are laid out [row][K]-contiguous so LDS tile staging uses the
// CDNA5 async path (global_load_async_to_lds_b128 + s_wait_asynccnt) with
// explicit double buffering.  Fixed problem dims are compile-time constants
// (epilogue scatters are shifts/masks, no integer division sequences).
// Workspace requirement ~1.3 GB.
// ---------------------------------------------------------------------------

// Fixed problem geometry (from the reference): B=32, T=512, E=512, H=8.
#define TT   512
#define EE   512
#define HH   8
#define TSH  9      // log2(TT)
#define HSH  3      // log2(HH)

typedef __bf16 bf16_t;
typedef __attribute__((ext_vector_type(16))) __bf16 v16bf;
typedef __attribute__((ext_vector_type(8)))  __bf16 v8bf;
typedef __attribute__((ext_vector_type(8)))  float  v8f;

static __device__ __forceinline__ bf16_t f2bf(float f) {
  unsigned u = __builtin_bit_cast(unsigned, f);
  unsigned r = (u + 0x7FFFu + ((u >> 16) & 1u)) >> 16;   // round-to-nearest-even
  unsigned short s = (unsigned short)r;
  return __builtin_bit_cast(bf16_t, s);
}

// CDNA5 async copy: 16 bytes global -> LDS, tracked by ASYNCcnt.
// lds byte address = low 32 bits of the flat address.
static __device__ __forceinline__ void async_b128(uint32_t lds_addr,
                                                  const bf16_t* g) {
  asm volatile("global_load_async_to_lds_b128 %0, %1, off"
               :: "v"(lds_addr), "v"(g) : "memory");
}
static __device__ __forceinline__ void wait_async0() {
  asm volatile("s_wait_asynccnt 0" ::: "memory");
}

// ---------------------------------------------------------------------------
// Tiled fp32 [K][N] -> bf16 [N][K] transpose (weights, once per launch).
// ---------------------------------------------------------------------------
__global__ __launch_bounds__(256)
void transpose_to_bf16_kernel(const float* __restrict__ src,  // [K][N]
                              bf16_t* __restrict__ dst,       // [N][K]
                              int K, int N) {
  __shared__ float tile[32][33];
  const int kb = blockIdx.y * 32, nb = blockIdx.x * 32;
  const int tx = threadIdx.x & 31, ty = threadIdx.x >> 5;  // 32 x 8
#pragma unroll
  for (int i = ty; i < 32; i += 8)
    tile[i][tx] = src[(size_t)(kb + i) * N + nb + tx];
  __syncthreads();
#pragma unroll
  for (int i = ty; i < 32; i += 8)
    dst[(size_t)(nb + i) * K + kb + tx] = f2bf(tile[tx][i]);
}

// ---------------------------------------------------------------------------
// Embedding gather + sinusoidal positional encoding (sin on ALL positions,
// faithful to reference). Writes fp32 x and bf16 x.
// ---------------------------------------------------------------------------
__global__ void embed_pos_kernel(const int* __restrict__ tokens,
                                 const float* __restrict__ emb,
                                 float* __restrict__ xf,
                                 bf16_t* __restrict__ xb) {
  const int row = blockIdx.x;            // b*T + t
  const int t   = row & (TT - 1);
  const int tok = tokens[row];
  for (int e = threadIdx.x; e < EE; e += blockDim.x) {
    float inv = __powf(10000.0f, -(float)(2 * (e >> 1)) / (float)EE);
    float v = emb[(size_t)tok * EE + e] + __sinf((float)t * inv);
    xf[(size_t)row * EE + e] = v;
    xb[(size_t)row * EE + e] = f2bf(v);
  }
}

// ---------------------------------------------------------------------------
// WMMA GEMM engine.  C[M,N] = A[M,K](bf16) * Bt[N,K](bf16)^T (+bias)(+relu)
// Both operands are [row][K] row-major -> all tile staging is contiguous
// 2x b128 async copies per thread, double-buffered in LDS.
// Block tile 128x128x32, 256 threads = 8 waves as 4x2; each wave owns a
// 32x64 tile = 2x4 v_wmma_f32_16x16x32_bf16 accumulators.
//
// MODE 0: standard row-major out (fp32 and/or bf16)
// MODE 1: QKV proj (Q,K); scatter bf16 out head-major [B,H,T,E]
// MODE 2: scores per head z=b*H+h: A=Q_head [T,E], Bt=K_head [T,E] (= S=QK^T)
// MODE 3: P.V per head: A=P[z,T,T], Bt=V_head^T [E,T]; bf16 out scattered
//         to [B,T, e*H+h] (reference's head-last interleave)
// MODE 4: V proj; scatter bf16 out transposed head-major [B,H,E,T]
// ---------------------------------------------------------------------------
#define BM 128
#define BN 128
#define BK 32
#define LDP 40   // LDS row pitch in bf16 elems (pad vs. bank conflicts)

template <int MODE>
__global__ __launch_bounds__(256)
void gemm_bf16_wmma(const bf16_t* __restrict__ A,
                    const bf16_t* __restrict__ Bt,
                    const float* __restrict__ bias,
                    float* __restrict__ outF,
                    bf16_t* __restrict__ outB,
                    int M, int N, int K, int lda, int ldb, int relu) {
  const int bz = blockIdx.z;
  if (MODE == 2) {
    A    += (size_t)bz * TT * EE;      // Q head
    Bt   += (size_t)bz * TT * EE;      // K head [s][e]
    outF += (size_t)bz * TT * TT;      // scores
  }
  if (MODE == 3) {
    A  += (size_t)bz * TT * TT;        // P head
    Bt += (size_t)bz * EE * TT;        // V head transposed [e][s]
  }
  const int m0 = blockIdx.y * BM, n0 = blockIdx.x * BN;
  const int tid = threadIdx.x, lane = tid & 31, wid = tid >> 5;
  const int wrow = (wid >> 1) * 32, wcol = (wid & 1) * 64;

  __shared__ __align__(16) bf16_t lA[2][BM * LDP];  // [buf][row][k]
  __shared__ __align__(16) bf16_t lB[2][BN * LDP];  // [buf][col][k]

  v8f acc[2][4] = {};

  // Unified contiguous loader: thread -> (row = tid/2, 16-elem half = tid&1).
  const int arow = tid >> 1, ah = (tid & 1) * 16;
  const bf16_t* pa = A + (size_t)(m0 + arow) * lda + ah;
  const bf16_t* pb = Bt + (size_t)(n0 + arow) * ldb + ah;
  const uint32_t la0 = (uint32_t)(uintptr_t)&lA[0][arow * LDP + ah];
  const uint32_t lb0 = (uint32_t)(uintptr_t)&lB[0][arow * LDP + ah];
  const uint32_t bufB = (uint32_t)(BM * LDP * sizeof(bf16_t));  // 10240

  // prologue: stage tile 0
  async_b128(la0, pa);       async_b128(la0 + 16, pa + 8);
  async_b128(lb0, pb);       async_b128(lb0 + 16, pb + 8);
  wait_async0();
  __syncthreads();

  int cur = 0;
  for (int k0 = 0; k0 < K; k0 += BK) {
    const bool more = (k0 + BK) < K;
    if (more) {
      const uint32_t bo = (cur ^ 1) ? bufB : 0u;
      async_b128(la0 + bo,      pa + k0 + BK);
      async_b128(la0 + bo + 16, pa + k0 + BK + 8);
      async_b128(lb0 + bo,      pb + k0 + BK);
      async_b128(lb0 + bo + 16, pb + k0 + BK + 8);
      if (k0 + 2 * BK < K) {
        __builtin_prefetch(pa + k0 + 2 * BK, 0, 1);
        __builtin_prefetch(pb + k0 + 2 * BK, 0, 1);
      }
    }

    // ---- fragments per ISA 16-bit A layout / symmetric B layout ----
    const bf16_t* bA = &lA[cur][0];
    const bf16_t* bB = &lB[cur][0];
    const int r15 = lane & 15, kh = lane >> 4;
    v16bf af[2], bfv[4];
#pragma unroll
    for (int i = 0; i < 2; i++) {
      const int row = wrow + i * 16 + r15;
      v8bf lo = *(const v8bf*)&bA[row * LDP + kh * 8];        // K = kh*8 .. +7
      v8bf hi = *(const v8bf*)&bA[row * LDP + 16 + kh * 8];   // K = 16+kh*8 ..
      af[i] = __builtin_shufflevector(lo, hi, 0, 1, 2, 3, 4, 5, 6, 7,
                                      8, 9, 10, 11, 12, 13, 14, 15);
    }
#pragma unroll
    for (int j = 0; j < 4; j++) {
      const int col = wcol + j * 16 + r15;
      v8bf lo = *(const v8bf*)&bB[col * LDP + kh * 8];        // K = kh*8 .. +7
      v8bf hi = *(const v8bf*)&bB[col * LDP + 16 + kh * 8];   // K = 16+kh*8 ..
      bfv[j] = __builtin_shufflevector(lo, hi, 0, 1, 2, 3, 4, 5, 6, 7,
                                       8, 9, 10, 11, 12, 13, 14, 15);
    }
#pragma unroll
    for (int i = 0; i < 2; i++)
#pragma unroll
      for (int j = 0; j < 4; j++)
        acc[i][j] = __builtin_amdgcn_wmma_f32_16x16x32_bf16(
            false, af[i], false, bfv[j], (short)0, acc[i][j], false, false);

    if (more) wait_async0();   // next buffer staged
    __syncthreads();           // everyone done reading cur
    cur ^= 1;
  }

  // ---- epilogue: C/D layout -> M = r + 8*(lane>=16), N = lane&15 ----
  const int r15 = lane & 15, kh = lane >> 4;
#pragma unroll
  for (int i = 0; i < 2; i++)
#pragma unroll
    for (int j = 0; j < 4; j++)
#pragma unroll
      for (int r = 0; r < 8; r++) {
        const int gm = m0 + wrow + i * 16 + r + 8 * kh;
        const int gn = n0 + wcol + j * 16 + r15;
        float v = acc[i][j][r];
        if (bias) v += bias[gn];
        if (relu) v = fmaxf(v, 0.0f);
        if (MODE == 0 || MODE == 2) {
          if (outF) outF[(size_t)gm * N + gn] = v;
          if (outB) outB[(size_t)gm * N + gn] = f2bf(v);
        } else if (MODE == 1) {
          const int b_ = gm >> TSH, t = gm & (TT - 1);
          const int e = gn >> HSH, h = gn & (HH - 1);
          outB[(((size_t)(b_ * HH + h) << TSH) + t) * EE + e] = f2bf(v);
        } else if (MODE == 4) {
          const int b_ = gm >> TSH, t = gm & (TT - 1);
          const int e = gn >> HSH, h = gn & (HH - 1);
          outB[((((size_t)(b_ * HH + h)) * EE + e) << TSH) + t] = f2bf(v);
        } else {  // MODE 3
          const int b_ = bz >> HSH, h = bz & (HH - 1);
          outB[((size_t)((b_ << TSH) + gm)) * (EE * HH) +
               ((size_t)gn << HSH) + h] = f2bf(v);
        }
      }
}

// ---------------------------------------------------------------------------
// Row softmax over scores [BH*T, T]: scale by 1/sqrt(E), optional pad mask
// (tokens==0 -> -1e9), write probabilities as bf16.  One wave32 per row.
// ---------------------------------------------------------------------------
__global__ __launch_bounds__(256)
void softmax_rows_kernel(const float* __restrict__ S, bf16_t* __restrict__ P,
                         const int* __restrict__ tokens,  // null => no mask
                         float scale) {
  const int row = blockIdx.x * 8 + (threadIdx.x >> 5);
  const int lane = threadIdx.x & 31;
  const int b_ = row >> (TSH + HSH);   // (row/T)/H
  const float* s = S + (size_t)row * TT;
  const int n = TT >> 5;  // 16
  float vals[16];
  float mx = -3.0e38f;
#pragma unroll
  for (int i = 0; i < n; i++) {
    const int c = lane + i * 32;
    float v = s[c] * scale;
    if (tokens && tokens[(b_ << TSH) + c] == 0) v -= 1e9f;
    vals[i] = v;
    mx = fmaxf(mx, v);
  }
  for (int o = 16; o; o >>= 1) mx = fmaxf(mx, __shfl_xor(mx, o, 32));
  float sum = 0.0f;
#pragma unroll
  for (int i = 0; i < n; i++) { vals[i] = __expf(vals[i] - mx); sum += vals[i]; }
  for (int o = 16; o; o >>= 1) sum += __shfl_xor(sum, o, 32);
  const float inv = 1.0f / sum;
  bf16_t* p = P + (size_t)row * TT;
#pragma unroll
  for (int i = 0; i < n; i++) p[lane + i * 32] = f2bf(vals[i] * inv);
}

// ---------------------------------------------------------------------------
// out = LayerNorm(X + U), eps=1e-6, gamma=1, beta=0.  One wave32 per row.
// ---------------------------------------------------------------------------
__global__ __launch_bounds__(256)
void add_layernorm_kernel(const float* __restrict__ X,
                          const float* __restrict__ U,
                          float* __restrict__ outF, bf16_t* __restrict__ outB) {
  const int row = blockIdx.x * 8 + (threadIdx.x >> 5);
  const int lane = threadIdx.x & 31;
  const float* x = X + (size_t)row * EE;
  const float* u = U + (size_t)row * EE;
  const int n = EE >> 5;  // 16
  float v[16];
  float sum = 0.0f;
#pragma unroll
  for (int i = 0; i < n; i++) { v[i] = x[lane + i * 32] + u[lane + i * 32]; sum += v[i]; }
  for (int o = 16; o; o >>= 1) sum += __shfl_xor(sum, o, 32);
  const float mu = sum * (1.0f / (float)EE);
  float var = 0.0f;
#pragma unroll
  for (int i = 0; i < n; i++) { const float d = v[i] - mu; var += d * d; }
  for (int o = 16; o; o >>= 1) var += __shfl_xor(var, o, 32);
  const float rs = rsqrtf(var * (1.0f / (float)EE) + 1e-6f);
#pragma unroll
  for (int i = 0; i < n; i++) {
    const float o = (v[i] - mu) * rs;
    outF[(size_t)row * EE + lane + i * 32] = o;
    outB[(size_t)row * EE + lane + i * 32] = f2bf(o);
  }
}

// ---------------------------------------------------------------------------
// avg over T, logits = avg @ Wc + bc.  One block per batch element.
// ---------------------------------------------------------------------------
__global__ __launch_bounds__(256)
void pool_logits_kernel(const float* __restrict__ X, const float* __restrict__ Wc,
                        const float* __restrict__ bc, float* __restrict__ out) {
  const int b = blockIdx.x, tid = threadIdx.x;
  __shared__ float avg[EE];
  __shared__ float red[256];
  for (int e = tid; e < EE; e += 256) {
    float s = 0.0f;
    for (int t = 0; t < TT; t++) s += X[((size_t)((b << TSH) + t)) * EE + e];
    avg[e] = s * (1.0f / (float)TT);
  }
  __syncthreads();
  for (int c = 0; c < 2; c++) {
    float p = 0.0f;
    for (int e = tid; e < EE; e += 256) p += avg[e] * Wc[e * 2 + c];
    red[tid] = p;
    __syncthreads();
    for (int s = 128; s; s >>= 1) { if (tid < s) red[tid] += red[tid + s]; __syncthreads(); }
    if (tid == 0) out[b * 2 + c] = red[0] + bc[c];
    __syncthreads();
  }
}

// loss = mean_b -sum(targets * log_softmax(logits)); single wave32.
__global__ void loss_kernel(const float* __restrict__ logits,
                            const float* __restrict__ targets,
                            float* __restrict__ out, int B) {
  const int lane = threadIdx.x;
  float l = 0.0f;
  if (lane < B) {
    const float l0 = logits[lane * 2], l1 = logits[lane * 2 + 1];
    const float m = fmaxf(l0, l1);
    const float lse = m + __logf(__expf(l0 - m) + __expf(l1 - m));
    l = -(targets[lane * 2] * (l0 - lse) + targets[lane * 2 + 1] * (l1 - lse));
  }
  for (int o = 16; o; o >>= 1) l += __shfl_xor(l, o, 32);
  if (lane == 0) out[B * 2] = l / (float)B;
}

// ---------------------------------------------------------------------------
extern "C" void kernel_launch(void* const* d_in, const int* in_sizes, int n_in,
                              void* d_out, int out_size, void* d_ws, size_t ws_size,
                              hipStream_t stream) {
  (void)in_sizes; (void)n_in; (void)out_size; (void)ws_size;
  const int Bb = 32, T = TT, E = EE, H = HH, EH = EE * HH, MLPN = 2048;
  const int Mrows = Bb * T;                    // 16384
  const float scale = 0.044194173824159216f;   // 1/sqrt(512)

  const int*   tokens  = (const int*)d_in[0];
  const float* targets = (const float*)d_in[1];
  const float* emb     = (const float*)d_in[2];
  const float *Wq[3], *Wk[3], *Wv[3], *Wu[3], *bu[3], *W1[3], *b1[3], *W2[3], *b2[3];
  for (int l = 0; l < 3; l++) {
    const int base = 3 + l * 9;
    Wq[l] = (const float*)d_in[base + 0];
    Wk[l] = (const float*)d_in[base + 1];
    Wv[l] = (const float*)d_in[base + 2];
    Wu[l] = (const float*)d_in[base + 3];
    bu[l] = (const float*)d_in[base + 4];
    W1[l] = (const float*)d_in[base + 5];
    b1[l] = (const float*)d_in[base + 6];
    W2[l] = (const float*)d_in[base + 7];
    b2[l] = (const float*)d_in[base + 8];
  }
  const float* Wc = (const float*)d_in[30];
  const float* bc = (const float*)d_in[31];
  float* out = (float*)d_out;

  // ---- workspace carve-up ----
  char* ws = (char*)d_ws;
  size_t off = 0;
  auto alloc = [&](size_t bytes) -> void* {
    void* p = ws + off;
    off += (bytes + 255) & ~(size_t)255;
    return p;
  };
  // bf16 TRANSPOSED weights [N][K]
  bf16_t *wqT[3], *wkT[3], *wvT[3], *wuT[3], *w1T[3], *w2T[3];
  for (int l = 0; l < 3; l++) {
    wqT[l] = (bf16_t*)alloc((size_t)E * EH * 2);
    wkT[l] = (bf16_t*)alloc((size_t)E * EH * 2);
    wvT[l] = (bf16_t*)alloc((size_t)E * EH * 2);
    wuT[l] = (bf16_t*)alloc((size_t)EH * E * 2);
    w1T[l] = (bf16_t*)alloc((size_t)E * MLPN * 2);
    w2T[l] = (bf16_t*)alloc((size_t)MLPN * E * 2);
  }
  float*  xf    = (float*)alloc((size_t)Mrows * E * 4);
  bf16_t* xb    = (bf16_t*)alloc((size_t)Mrows * E * 2);
  bf16_t* qh    = (bf16_t*)alloc((size_t)Bb * H * T * E * 2);  // [B,H,T,E]
  bf16_t* kh    = (bf16_t*)alloc((size_t)Bb * H * T * E * 2);  // [B,H,T,E]
  bf16_t* vht   = (bf16_t*)alloc((size_t)Bb * H * E * T * 2);  // [B,H,E,T]
  float*  sc    = (float*)alloc((size_t)Bb * H * T * T * 4);   // scores
  bf16_t* pm    = (bf16_t*)alloc((size_t)Bb * H * T * T * 2);  // probs
  bf16_t* wvo   = (bf16_t*)alloc((size_t)Mrows * EH * 2);      // att output
  float*  unif  = (float*)alloc((size_t)Mrows * E * 4);
  float*  nmF   = (float*)alloc((size_t)Mrows * E * 4);
  bf16_t* nmB   = (bf16_t*)alloc((size_t)Mrows * E * 2);
  bf16_t* h1b   = (bf16_t*)alloc((size_t)Mrows * MLPN * 2);
  float*  h2f   = (float*)alloc((size_t)Mrows * E * 4);

  // weight transposes (fp32 [K][N] -> bf16 [N][K]), once per launch
  auto tr = [&](const float* s, bf16_t* d, int K_, int N_) {
    transpose_to_bf16_kernel<<<dim3(N_ / 32, K_ / 32), 256, 0, stream>>>(s, d, K_, N_);
  };
  for (int l = 0; l < 3; l++) {
    tr(Wq[l], wqT[l], E, EH);
    tr(Wk[l], wkT[l], E, EH);
    tr(Wv[l], wvT[l], E, EH);
    tr(Wu[l], wuT[l], EH, E);
    tr(W1[l], w1T[l], E, MLPN);
    tr(W2[l], w2T[l], MLPN, E);
  }

  embed_pos_kernel<<<Mrows, 256, 0, stream>>>(tokens, emb, xf, xb);

  for (int l = 0; l < 3; l++) {
    // Q/K -> head-major [B,H,T,E]; V -> transposed head-major [B,H,E,T]
    dim3 gq(EH / BN, Mrows / BM, 1);
    gemm_bf16_wmma<1><<<gq, 256, 0, stream>>>(xb, wqT[l], nullptr, nullptr, qh,
                                              Mrows, EH, E, E, E, 0);
    gemm_bf16_wmma<1><<<gq, 256, 0, stream>>>(xb, wkT[l], nullptr, nullptr, kh,
                                              Mrows, EH, E, E, E, 0);
    gemm_bf16_wmma<4><<<gq, 256, 0, stream>>>(xb, wvT[l], nullptr, nullptr, vht,
                                              Mrows, EH, E, E, E, 0);
    // scores = Q K^T / sqrt(E) per head
    dim3 gs(T / BN, T / BM, Bb * H);
    gemm_bf16_wmma<2><<<gs, 256, 0, stream>>>(qh, kh, nullptr, sc, nullptr,
                                              T, T, E, E, E, 0);
    softmax_rows_kernel<<<(Bb * H * T) / 8, 256, 0, stream>>>(
        sc, pm, (l == 0) ? tokens : nullptr, scale);
    // wv = P V per head -> [B,T, e*H+h] bf16
    dim3 gp(E / BN, T / BM, Bb * H);
    gemm_bf16_wmma<3><<<gp, 256, 0, stream>>>(pm, vht, nullptr, nullptr, wvo,
                                              T, E, T, T, T, 0);
    // unified = wv @ Wu + bu
    dim3 gu(E / BN, Mrows / BM, 1);
    gemm_bf16_wmma<0><<<gu, 256, 0, stream>>>(wvo, wuT[l], bu[l], unif, nullptr,
                                              Mrows, E, EH, EH, EH, 0);
    add_layernorm_kernel<<<Mrows / 8, 256, 0, stream>>>(xf, unif, nmF, nmB);
    // MLP
    dim3 g1(MLPN / BN, Mrows / BM, 1);
    gemm_bf16_wmma<0><<<g1, 256, 0, stream>>>(nmB, w1T[l], b1[l], nullptr, h1b,
                                              Mrows, MLPN, E, E, E, 1);
    dim3 g2(E / BN, Mrows / BM, 1);
    gemm_bf16_wmma<0><<<g2, 256, 0, stream>>>(h1b, w2T[l], b2[l], h2f, nullptr,
                                              Mrows, E, MLPN, MLPN, MLPN, 0);
    add_layernorm_kernel<<<Mrows / 8, 256, 0, stream>>>(nmF, h2f, xf, xb);
  }

  pool_logits_kernel<<<Bb, 256, 0, stream>>>(xf, Wc, bc, out);
  loss_kernel<<<1, 32, 0, stream>>>(out, targets, out, Bb);
}
